// RelativePositionSelfAttention_64914135712469
// MI455X (gfx1250) — matrix operand order repaired and checked by reading, not compiled, not run on hardware
//
#include <hip/hip_runtime.h>
#include <hip/hip_bf16.h>

typedef __bf16 bf16_t;
typedef __attribute__((ext_vector_type(16))) __bf16 v16bf;
typedef __attribute__((ext_vector_type(8)))  __bf16 v8bf;
typedef __attribute__((ext_vector_type(8)))  float  v8f;

#define BATCH  8
#define SEQ    512
#define DMODEL 1024
#define NHEAD  16
#define DK     64
#define BH     (BATCH * NHEAD)   // 128
#define MROWS  (BATCH * SEQ)     // 4096
#define RELROWS (2 * SEQ - 1)    // 1023

// ---------- WMMA helpers (contiguous K per lane half, same in A and B) ----------

// A operand, 16-bit, 16x32: row = lane&15, 16 consecutive K starting at k0 + 16*(lane>=16).
static __device__ inline v16bf frag_a_bf16(const bf16_t* base, int ld, int m0, int k0, int lane) {
    const int half = lane >> 4;
    const bf16_t* p = base + (size_t)(m0 + (lane & 15)) * ld + k0 + half * 16;
    v16bf f;
    ((v8bf*)&f)[0] = *(const v8bf*)(p);
    ((v8bf*)&f)[1] = *(const v8bf*)(p + 8);
    return f;
}

// B operand, 16-bit, 32x16: col = lane&15, 16 consecutive K starting at k0 + 16*(lane>=16).
// base is [N, K] row-major ("transposed" storage).
static __device__ inline v16bf frag_b_bf16(const bf16_t* base, int ld, int n0, int k0, int lane) {
    const bf16_t* p = base + (size_t)(n0 + (lane & 15)) * ld + k0 + (lane >> 4) * 16;
    v16bf f;
    ((v8bf*)&f)[0] = *(const v8bf*)(p);
    ((v8bf*)&f)[1] = *(const v8bf*)(p + 8);
    return f;
}

static __device__ inline v8f wmma_bf16(v16bf a, v16bf b, v8f c) {
    return __builtin_amdgcn_wmma_f32_16x16x32_bf16(false, a, false, b, (short)0, c, false, false);
}

// ---------- conversion kernels ----------

__global__ __launch_bounds__(256) void cvt_bf16_kernel(const float* __restrict__ in,
                                                       bf16_t* __restrict__ out, int n) {
    int i = blockIdx.x * 256 + threadIdx.x;
    if (i < n) out[i] = (bf16_t)in[i];
}

// Wt[n][k] = W[k][n], 1024x1024
__global__ __launch_bounds__(256) void transpose_bf16_kernel(const float* __restrict__ W,
                                                             bf16_t* __restrict__ Wt) {
    int idx = blockIdx.x * 256 + threadIdx.x;           // 0 .. 1M-1
    int n = idx >> 10, k = idx & 1023;
    Wt[idx] = (bf16_t)W[(size_t)k * DMODEL + n];
}

// ---------- GEMM: C[M,N] = A[M,K](bf16) * Bt[N,K](bf16) + bias ----------
// OUTMODE 0: bf16 scatter to Q/K layout [b,h,l,d]
// OUTMODE 1: bf16 scatter to V-transposed layout [b,h,d,l]
// OUTMODE 2: f32 plain [M,N]
template<int OUTMODE>
__global__ __launch_bounds__(256) void gemm_bias_kernel(
    const bf16_t* __restrict__ A, const bf16_t* __restrict__ Bt,
    const float* __restrict__ bias, void* __restrict__ outp,
    int M, int N, int K)
{
    const int lane = threadIdx.x & 31;
    const int wave = threadIdx.x >> 5;                  // 8 waves: 2 (M) x 4 (N)
    const int m0 = blockIdx.y * 128 + (wave & 1) * 64;
    const int n0 = blockIdx.x * 256 + (wave >> 1) * 64;

    v8f acc[4][4];
    #pragma unroll
    for (int mi = 0; mi < 4; ++mi)
        #pragma unroll
        for (int ni = 0; ni < 4; ++ni) acc[mi][ni] = {};

    for (int k0 = 0; k0 < K; k0 += 32) {
        v16bf af[4], bf_[4];
        #pragma unroll
        for (int mi = 0; mi < 4; ++mi) af[mi]  = frag_a_bf16(A,  K, m0 + mi * 16, k0, lane);
        #pragma unroll
        for (int ni = 0; ni < 4; ++ni) bf_[ni] = frag_b_bf16(Bt, K, n0 + ni * 16, k0, lane);
        #pragma unroll
        for (int mi = 0; mi < 4; ++mi)
            #pragma unroll
            for (int ni = 0; ni < 4; ++ni)
                acc[mi][ni] = wmma_bf16(af[mi], bf_[ni], acc[mi][ni]);
    }

    const int mh = (lane >> 4) * 8;
    #pragma unroll
    for (int mi = 0; mi < 4; ++mi)
        #pragma unroll
        for (int ni = 0; ni < 4; ++ni)
            #pragma unroll
            for (int r = 0; r < 8; ++r) {
                int m = m0 + mi * 16 + r + mh;
                int n = n0 + ni * 16 + (lane & 15);
                float v = acc[mi][ni][r] + bias[n];
                if (OUTMODE == 2) {
                    ((float*)outp)[(size_t)m * N + n] = v;
                } else if (OUTMODE == 0) {
                    // [b,h,l,d]:  b=m>>9, l=m&511, h=n>>6, d=n&63
                    size_t idx = (((size_t)((m >> 9) * NHEAD + (n >> 6)) * SEQ) + (m & 511)) * DK + (n & 63);
                    ((bf16_t*)outp)[idx] = (bf16_t)v;
                } else {
                    // [b,h,d,l]
                    size_t idx = (((size_t)((m >> 9) * NHEAD + (n >> 6)) * DK) + (n & 63)) * SEQ + (m & 511);
                    ((bf16_t*)outp)[idx] = (bf16_t)v;
                }
            }
}

// ---------- scores: S = 0.125 * Q K^T + skewed rel band (one wave / 64x64 tile) ----------
__global__ __launch_bounds__(32) void attn_score_kernel(
    const bf16_t* __restrict__ Qb,   // [BH, SEQ, DK]
    const bf16_t* __restrict__ Kb,   // [BH, SEQ, DK]  (B-layout for QK^T)
    const bf16_t* __restrict__ relb, // [1023, DK]
    float* __restrict__ S)           // [BH, SEQ, SEQ]
{
    __shared__ float band[64][132];  // 127-wide rel band, padded
    const int lane = threadIdx.x & 31;
    const int j0 = blockIdx.x * 64, i0 = blockIdx.y * 64, bh = blockIdx.z;
    const bf16_t* Qs = Qb + (size_t)bh * SEQ * DK;
    const bf16_t* Ks = Kb + (size_t)bh * SEQ * DK;
    float* Ss = S + (size_t)bh * SEQ * SEQ;
    const int pbase = j0 - i0 + 448;           // min rel index used by this tile
    const int mh = (lane >> 4) * 8;

    // Phase A: band[0:64, 0:128] = Q(64xDK) * rel[pbase : pbase+128)^T  (via WMMA)
    for (int nt = 0; nt < 8; ++nt) {
        v8f bacc[4];
        #pragma unroll
        for (int mi = 0; mi < 4; ++mi) bacc[mi] = {};
        for (int ks = 0; ks < DK; ks += 32) {
            int prow = pbase + nt * 16 + (lane & 15);
            if (prow > RELROWS - 1) prow = RELROWS - 1;   // band edge: value never gathered
            const bf16_t* p = relb + (size_t)prow * DK + ks + (lane >> 4) * 16;
            v16bf bf_;
            ((v8bf*)&bf_)[0] = *(const v8bf*)(p);
            ((v8bf*)&bf_)[1] = *(const v8bf*)(p + 8);
            #pragma unroll
            for (int mi = 0; mi < 4; ++mi) {
                v16bf a = frag_a_bf16(Qs, DK, i0 + mi * 16, ks, lane);
                bacc[mi] = wmma_bf16(a, bf_, bacc[mi]);
            }
        }
        #pragma unroll
        for (int mi = 0; mi < 4; ++mi)
            #pragma unroll
            for (int r = 0; r < 8; ++r)
                band[mi * 16 + r + mh][nt * 16 + (lane & 15)] = bacc[mi][r];
    }
    __syncthreads();

    // Phase B: S tile = Q K^T, fused scale + skew-gather add
    v8f sacc[4][4];
    #pragma unroll
    for (int mi = 0; mi < 4; ++mi)
        #pragma unroll
        for (int ni = 0; ni < 4; ++ni) sacc[mi][ni] = {};
    for (int ks = 0; ks < DK; ks += 32) {
        v16bf aa[4], bb[4];
        #pragma unroll
        for (int mi = 0; mi < 4; ++mi) aa[mi] = frag_a_bf16(Qs, DK, i0 + mi * 16, ks, lane);
        #pragma unroll
        for (int ni = 0; ni < 4; ++ni) bb[ni] = frag_b_bf16(Ks, DK, j0 + ni * 16, ks, lane);
        #pragma unroll
        for (int mi = 0; mi < 4; ++mi)
            #pragma unroll
            for (int ni = 0; ni < 4; ++ni)
                sacc[mi][ni] = wmma_bf16(aa[mi], bb[ni], sacc[mi][ni]);
    }
    #pragma unroll
    for (int mi = 0; mi < 4; ++mi)
        #pragma unroll
        for (int ni = 0; ni < 4; ++ni)
            #pragma unroll
            for (int r = 0; r < 8; ++r) {
                int il = mi * 16 + r + mh;
                int jl = ni * 16 + (lane & 15);
                float v = sacc[mi][ni][r] * 0.125f + band[il][jl - il + 63]; // p = j-i+L-1
                Ss[(size_t)(i0 + il) * SEQ + j0 + jl] = v;
            }
}

// ---------- in-place row softmax: one wave per 512-wide row ----------
__global__ __launch_bounds__(256) void softmax_kernel(float* __restrict__ S) {
    const int lane = threadIdx.x & 31;
    const int wave = threadIdx.x >> 5;
    float* p = S + ((size_t)blockIdx.x * 8 + wave) * SEQ + lane * 16;
    float4 v[4];
    #pragma unroll
    for (int c = 0; c < 4; ++c) v[c] = *(const float4*)(p + c * 4);
    float mx = -3.0e38f;
    #pragma unroll
    for (int c = 0; c < 4; ++c)
        mx = fmaxf(mx, fmaxf(fmaxf(v[c].x, v[c].y), fmaxf(v[c].z, v[c].w)));
    #pragma unroll
    for (int off = 16; off > 0; off >>= 1) mx = fmaxf(mx, __shfl_xor(mx, off, 32));
    float sum = 0.f;
    #pragma unroll
    for (int c = 0; c < 4; ++c) {
        v[c].x = __expf(v[c].x - mx); v[c].y = __expf(v[c].y - mx);
        v[c].z = __expf(v[c].z - mx); v[c].w = __expf(v[c].w - mx);
        sum += v[c].x + v[c].y + v[c].z + v[c].w;
    }
    #pragma unroll
    for (int off = 16; off > 0; off >>= 1) sum += __shfl_xor(sum, off, 32);
    float inv = __frcp_rn(sum);
    #pragma unroll
    for (int c = 0; c < 4; ++c) {
        v[c].x *= inv; v[c].y *= inv; v[c].z *= inv; v[c].w *= inv;
        *(float4*)(p + c * 4) = v[c];
    }
}

// ---------- A * V: 8 waves, one 64-row tile each, full 64-wide N ----------
__global__ __launch_bounds__(256) void attn_av_kernel(
    const float*  __restrict__ A,   // [BH, SEQ, SEQ] probabilities (f32)
    const bf16_t* __restrict__ Vt,  // [BH, DK, SEQ]  (B-layout)
    bf16_t* __restrict__ Ctx)       // [MROWS, DMODEL]
{
    const int lane = threadIdx.x & 31;
    const int wave = threadIdx.x >> 5;
    const int bh = blockIdx.x;
    const int i0 = wave * 64;
    const float*  As = A  + (size_t)bh * SEQ * SEQ;
    const bf16_t* Vs = Vt + (size_t)bh * DK * SEQ;
    const int b = bh >> 4, h = bh & 15;
    const int mh = (lane >> 4) * 8;

    v8f acc[4][4];
    #pragma unroll
    for (int mi = 0; mi < 4; ++mi)
        #pragma unroll
        for (int ni = 0; ni < 4; ++ni) acc[mi][ni] = {};

    for (int k0 = 0; k0 < SEQ; k0 += 32) {
        v16bf aa[4], bb[4];
        #pragma unroll
        for (int mi = 0; mi < 4; ++mi) {
            const float* ap = As + (size_t)(i0 + mi * 16 + (lane & 15)) * SEQ + k0 + (lane >> 4) * 16;
            v16bf f;
            #pragma unroll
            for (int t = 0; t < 8; ++t) ((bf16_t*)&f)[t]     = (bf16_t)ap[t];
            #pragma unroll
            for (int t = 0; t < 8; ++t) ((bf16_t*)&f)[8 + t] = (bf16_t)ap[8 + t];
            aa[mi] = f;
        }
        #pragma unroll
        for (int ni = 0; ni < 4; ++ni) bb[ni] = frag_b_bf16(Vs, SEQ, ni * 16, k0, lane);
        #pragma unroll
        for (int mi = 0; mi < 4; ++mi)
            #pragma unroll
            for (int ni = 0; ni < 4; ++ni)
                acc[mi][ni] = wmma_bf16(aa[mi], bb[ni], acc[mi][ni]);
    }
    #pragma unroll
    for (int mi = 0; mi < 4; ++mi)
        #pragma unroll
        for (int ni = 0; ni < 4; ++ni)
            #pragma unroll
            for (int r = 0; r < 8; ++r) {
                int i = i0 + mi * 16 + r + mh;
                int d = ni * 16 + (lane & 15);
                Ctx[(size_t)(b * SEQ + i) * DMODEL + h * DK + d] = (bf16_t)acc[mi][ni][r];
            }
}

// ---------- host launcher ----------
extern "C" void kernel_launch(void* const* d_in, const int* in_sizes, int n_in,
                              void* d_out, int out_size, void* d_ws, size_t ws_size,
                              hipStream_t stream) {
    (void)in_sizes; (void)n_in; (void)out_size; (void)ws_size;
    const float* x    = (const float*)d_in[0];
    const float* Wq   = (const float*)d_in[1];
    const float* bq   = (const float*)d_in[2];
    const float* Wk   = (const float*)d_in[3];
    const float* bk   = (const float*)d_in[4];
    const float* Wv   = (const float*)d_in[5];
    const float* bv   = (const float*)d_in[6];
    const float* Wfc  = (const float*)d_in[7];
    const float* bfc  = (const float*)d_in[8];
    const float* rel  = (const float*)d_in[9];
    // d_in[10] = patch_index == -1 (static branch in reference): series == A

    float* out = (float*)d_out;                              // [8,512,1024]
    float* S   = out + (size_t)MROWS * DMODEL;               // [8,16,512,512] (series region)

    uint8_t* ws = (uint8_t*)d_ws;
    bf16_t* Xb   = (bf16_t*)ws;                               ws += (size_t)MROWS * DMODEL * 2;   // 8 MB
    bf16_t* Wqt  = (bf16_t*)ws;                               ws += (size_t)DMODEL * DMODEL * 2;  // 2 MB
    bf16_t* Wkt  = (bf16_t*)ws;                               ws += (size_t)DMODEL * DMODEL * 2;
    bf16_t* Wvt  = (bf16_t*)ws;                               ws += (size_t)DMODEL * DMODEL * 2;
    bf16_t* Wft  = (bf16_t*)ws;                               ws += (size_t)DMODEL * DMODEL * 2;
    bf16_t* relb = (bf16_t*)ws;                               ws += 131072;                       // 1023*64*2 padded
    bf16_t* Qb   = (bf16_t*)ws;                               ws += (size_t)BH * SEQ * DK * 2;    // 8 MB
    bf16_t* Kb   = (bf16_t*)ws;                               ws += (size_t)BH * SEQ * DK * 2;
    bf16_t* Vtb  = (bf16_t*)ws;                               ws += (size_t)BH * DK * SEQ * 2;
    bf16_t* Ctxb = (bf16_t*)ws;                               ws += (size_t)MROWS * DMODEL * 2;

    // 1) bf16 conversions (weights transposed into B-operand layout [N,K])
    cvt_bf16_kernel<<<(MROWS * DMODEL + 255) / 256, 256, 0, stream>>>(x, Xb, MROWS * DMODEL);
    transpose_bf16_kernel<<<(DMODEL * DMODEL) / 256, 256, 0, stream>>>(Wq,  Wqt);
    transpose_bf16_kernel<<<(DMODEL * DMODEL) / 256, 256, 0, stream>>>(Wk,  Wkt);
    transpose_bf16_kernel<<<(DMODEL * DMODEL) / 256, 256, 0, stream>>>(Wv,  Wvt);
    transpose_bf16_kernel<<<(DMODEL * DMODEL) / 256, 256, 0, stream>>>(Wfc, Wft);
    cvt_bf16_kernel<<<(RELROWS * DK + 255) / 256, 256, 0, stream>>>(rel, relb, RELROWS * DK);

    // 2) Q / K / V projections (WMMA), scattered into attention layouts
    dim3 gGemm(DMODEL / 256, MROWS / 128);  // (4, 32)
    gemm_bias_kernel<0><<<gGemm, 256, 0, stream>>>(Xb, Wqt, bq, Qb,  MROWS, DMODEL, DMODEL);
    gemm_bias_kernel<0><<<gGemm, 256, 0, stream>>>(Xb, Wkt, bk, Kb,  MROWS, DMODEL, DMODEL);
    gemm_bias_kernel<1><<<gGemm, 256, 0, stream>>>(Xb, Wvt, bv, Vtb, MROWS, DMODEL, DMODEL);

    // 3) scores with fused relative-position band (writes series region)
    attn_score_kernel<<<dim3(SEQ / 64, SEQ / 64, BH), 32, 0, stream>>>(Qb, Kb, relb, S);

    // 4) in-place softmax -> A (this IS the `series` output)
    softmax_kernel<<<(size_t)BH * SEQ / 8, 256, 0, stream>>>(S);

    // 5) context = A * V (WMMA)
    attn_av_kernel<<<BH, 256, 0, stream>>>(S, Vtb, Ctxb);

    // 6) output projection (f32 out)
    gemm_bias_kernel<2><<<gGemm, 256, 0, stream>>>(Ctxb, Wft, bfc, out, MROWS, DMODEL, DMODEL);
}
